// Asymm_3d_spconv_15968688407226
// MI455X (gfx1250) — matrix operand order repaired and checked by reading, not compiled, not run on hardware
//
#include <hip/hip_runtime.h>

// ---------------------------------------------------------------------------
// CDNA5 (gfx1250, wave32) Asymm-3D-spconv U-Net.
// Every conv is an implicit GEMM on v_wmma_f32_16x16x32_f16:
//   M = 16 output voxels, N = 2 x 16 output channels, K = taps*Cin (chunks 32).
// Activations are stored f16 (whole net ~105MB -> L2-resident on the 192MB
// MI455X L2; HBM@23.3TB/s is never the bottleneck), weights pre-transformed
// to f16 [tap][co_pad32][ci] (spatial flip folded in for transposed convs).
// Inner loop per two 16x16 output tiles: 2x b128 A-loads + 4x b128 B-loads +
// 2x WMMA, no converts, no integer division. Epilogue (bias, mask, BN scale,
// lrelu/sigmoid, skip-add) is fused into the conv store in fp32.
// ---------------------------------------------------------------------------

typedef __attribute__((ext_vector_type(16))) _Float16 v16h;
typedef __attribute__((ext_vector_type(8)))  _Float16 v8h;
typedef __attribute__((ext_vector_type(8)))  float    v8f;

#define BN_SCALE 0.99999499998749993f /* 1/sqrt(1+1e-5) */

// ---------------------------------------------------------------------------
// Templated masked conv3d (+ lhs-dilation for transposed conv; dilations here
// are only 1 or 2). Dual input: channel ci reads `in` if ci < CSPLIT else
// `in2` (virtual concat for the logits layer).
// actmode: 0 = none, 1 = leaky_relu(0.01), 2 = sigmoid.
// store: v = (acc + bias) * mask * scale; v = act(v); v += addbuf
// ---------------------------------------------------------------------------
template <int KD, int KH, int KW, int CI, int CSPLIT = CI>
__global__ __launch_bounds__(32) void conv_wmma_t(
    const _Float16* __restrict__ in, const _Float16* __restrict__ in2,
    int Di, int Hi, int Wi,
    const _Float16* __restrict__ wt, int CoP,
    _Float16* __restrict__ out, int Do, int Ho, int Wo, int Co,
    int sz, int sy, int sx, int pz, int py, int px, int dz, int dy, int dx,
    const float* __restrict__ mask, const _Float16* __restrict__ addbuf,
    const float* __restrict__ bias, int actmode, float scale)
{
  constexpr int TAPS = KD * KH * KW;
  const int lane  = threadIdx.x;   // wave32
  const int g     = lane >> 4;     // lane-half group
  const int r16   = lane & 15;
  const int nvox  = Do * Ho * Wo;
  const int tileV = blockIdx.x * 16;
  const int coB   = blockIdx.y * 32;  // two 16-wide N tiles per wave
  const int n0c   = coB + r16;        // N column, tile 0 (n < CoP; B zero-padded)

  const int vm = tileV + r16;
  const bool vrow = (vm < nvox);
  int zo = 0, yo = 0, xo = 0;
  if (vrow) { xo = vm % Wo; int t = vm / Wo; yo = t % Ho; zo = t / Ho; }

  // per-tap spatial row index + validity (dilation values are only 1 or 2)
  auto rowspatial = [&](int tap, bool& ok) -> size_t {
    const int kw_ = tap % KW;
    const int kh_ = (tap / KW) % KH;
    const int kd_ = tap / (KW * KH);
    const int tz = zo * sz - pz + kd_;
    const int ty = yo * sy - py + kh_;
    const int tx = xo * sx - px + kw_;
    int zi = tz, yi = ty, xi = tx;
    bool o = vrow && tz >= 0 && ty >= 0 && tx >= 0;
    if (dz > 1) { o = o && ((tz & 1) == 0); zi = tz >> 1; }
    if (dy > 1) { o = o && ((ty & 1) == 0); yi = ty >> 1; }
    if (dx > 1) { o = o && ((tx & 1) == 0); xi = tx >> 1; }
    o = o && (zi < Di) && (yi < Hi) && (xi < Wi);
    ok = o;
    return ((size_t)zi * Hi + yi) * Wi + xi;
  };

  v8f acc0 = {0.f, 0.f, 0.f, 0.f, 0.f, 0.f, 0.f, 0.f};
  v8f acc1 = {0.f, 0.f, 0.f, 0.f, 0.f, 0.f, 0.f, 0.f};

  if constexpr (CI >= 32) {
#pragma unroll 1
    for (int tap = 0; tap < TAPS; ++tap) {
      bool ok;
      const size_t s = rowspatial(tap, ok);
      const _Float16* brow = wt + ((size_t)tap * CoP + n0c) * CI;
#pragma unroll
      for (int c0 = 0; c0 < CI; c0 += 32) {
        // A: h<8 -> ch c0+g*8+h ; h>=8 -> ch c0+16+g*8+(h-8)  (f16, contiguous)
        const _Float16* ap;
        if constexpr (CSPLIT == CI) {
          ap = in + s * (size_t)CI + c0;
        } else {
          ap = (c0 < CSPLIT) ? (in + s * (size_t)CSPLIT + c0)
                             : (in2 + s * (size_t)(CI - CSPLIT) + (c0 - CSPLIT));
        }
        v8h alo = {}, ahi = {};
        if (ok) {
          const _Float16* p = ap + g * 8;
          alo = *(const v8h*)(p);
          ahi = *(const v8h*)(p + 16);
        }
        v16h a;
#pragma unroll
        for (int i = 0; i < 8; ++i) { a[i] = alo[i]; a[8 + i] = ahi[i]; }
        // B: h<8 -> ch c0+g*8+h ; h>=8 -> ch c0+16+g*8+(h-8) ; tile1 is +16 columns = +16*CI halves
        const v8h b00 = *(const v8h*)(brow + c0 + g * 8);
        const v8h b01 = *(const v8h*)(brow + c0 + g * 8 + 16);
        const v8h b10 = *(const v8h*)(brow + 16 * CI + c0 + g * 8);
        const v8h b11 = *(const v8h*)(brow + 16 * CI + c0 + g * 8 + 16);
        v16h b0, b1;
#pragma unroll
        for (int i = 0; i < 8; ++i) {
          b0[i] = b00[i]; b0[8 + i] = b01[i];
          b1[i] = b10[i]; b1[8 + i] = b11[i];
        }
        acc0 = __builtin_amdgcn_wmma_f32_16x16x32_f16(
            false, a, false, b0, (short)0, acc0, false, false);
        acc1 = __builtin_amdgcn_wmma_f32_16x16x32_f16(
            false, a, false, b1, (short)0, acc1, false, false);
      }
    }
  } else {
    // CI == 16: two taps per K-chunk (taps padded to even in wt prep)
    constexpr int TP = (TAPS + 1) & ~1;
#pragma unroll 1
    for (int t0 = 0; t0 < TP; t0 += 2) {
      bool ok0 = false, ok1 = false;
      const size_t s0 = rowspatial(t0, ok0);
      size_t s1 = 0;
      if (t0 + 1 < TAPS) s1 = rowspatial(t0 + 1, ok1);
      v8h alo = {}, ahi = {};
      if (ok0) alo = *(const v8h*)(in + s0 * 16 + g * 8);
      if (ok1) ahi = *(const v8h*)(in + s1 * 16 + g * 8);
      v16h a;
#pragma unroll
      for (int i = 0; i < 8; ++i) { a[i] = alo[i]; a[8 + i] = ahi[i]; }
      // B: h<8 -> tap t0 ch g*8+h ; h>=8 -> tap t0+1 ch g*8+(h-8) (zero-padded taps)
      const _Float16* brow = wt + ((size_t)t0 * CoP + n0c) * 16 + g * 8;
      const v8h b00 = *(const v8h*)(brow);
      const v8h b01 = *(const v8h*)(brow + (size_t)CoP * 16);
      const v8h b10 = *(const v8h*)(brow + 256);  // +16 columns * 16 ch
      const v8h b11 = *(const v8h*)(brow + 256 + (size_t)CoP * 16);
      v16h b0, b1;
#pragma unroll
      for (int i = 0; i < 8; ++i) {
        b0[i] = b00[i]; b0[8 + i] = b01[i];
        b1[i] = b10[i]; b1[8 + i] = b11[i];
      }
      acc0 = __builtin_amdgcn_wmma_f32_16x16x32_f16(
          false, a, false, b0, (short)0, acc0, false, false);
      acc1 = __builtin_amdgcn_wmma_f32_16x16x32_f16(
          false, a, false, b1, (short)0, acc1, false, false);
    }
  }

  // fused epilogue + store (32-bit C/D 16x16: VGPR r -> M = r + 8*g, N = lane%16)
  auto store_tile = [&](const v8f& acc, int n) {
    if (n >= Co) return;
#pragma unroll
    for (int r = 0; r < 8; ++r) {
      const int vox = tileV + r + 8 * g;
      if (vox < nvox) {
        float v = acc[r];
        if (bias) v += bias[n];
        if (mask) v *= mask[vox];
        v *= scale;
        if      (actmode == 1) v = (v > 0.f) ? v : 0.01f * v;
        else if (actmode == 2) v = 1.f / (1.f + __expf(-v));
        if (addbuf) v += (float)addbuf[(size_t)vox * Co + n];
        out[(size_t)vox * Co + n] = (_Float16)v;
      }
    }
  };
  store_tile(acc0, n0c);
  store_tile(acc1, n0c + 16);
}

// ---------------------------------------------------------------------------
// Weight prep: fp32 [kd][kh][kw][ci][co] -> f16 [tap][co_pad32][ci],
// spatially flipped when flip!=0 (all-odd kernels => linear reversal),
// zero padding for co in [Co,CoP) and tap in [taps,tapsPad).
// ---------------------------------------------------------------------------
__global__ void wprep_kernel(const float* __restrict__ w, _Float16* __restrict__ wt,
                             int taps, int tapsPad, int Ci, int Co, int CoP, int flip) {
  const size_t i = (size_t)blockIdx.x * blockDim.x + threadIdx.x;
  const size_t total = (size_t)tapsPad * CoP * Ci;
  if (i >= total) return;
  const int ci = (int)(i % Ci);
  const size_t t = i / Ci;
  const int co = (int)(t % CoP);
  const int tap = (int)(t / CoP);
  float v = 0.f;
  if (tap < taps && co < Co) {
    const int ts = flip ? (taps - 1 - tap) : tap;
    v = w[((size_t)ts * Ci + ci) * Co + co];
  }
  wt[i] = (_Float16)v;
}

// ---------------------------------------------------------------------------
// Small helper kernels
// ---------------------------------------------------------------------------
__global__ void zero_kernel(float* __restrict__ p, size_t n) {
  size_t i = (size_t)blockIdx.x * blockDim.x + threadIdx.x;
  const size_t s = (size_t)gridDim.x * blockDim.x;
  for (; i < n; i += s) p[i] = 0.f;
}

__global__ void scatter_kernel(const float* __restrict__ feat,
                               const int* __restrict__ coors, int nv, int Cin,
                               _Float16* __restrict__ X, float* __restrict__ m0,
                               int Hh, int Ww) {
  const int i = blockIdx.x * blockDim.x + threadIdx.x;
  if (i >= nv * Cin) return;
  const int v = i / Cin, c = i - v * Cin;
  const int z = coors[v * 4 + 1], y = coors[v * 4 + 2], x = coors[v * 4 + 3];
  const size_t base = ((size_t)z * Hh + y) * Ww + x;
  X[base * Cin + c] = (_Float16)feat[(size_t)v * Cin + c];
  if (c == 0) m0[base] = 1.f;
}

__global__ void mask_down_kernel(const float* __restrict__ mi, int Di, int Hi, int Wi,
                                 float* __restrict__ mo, int Do, int Ho, int Wo,
                                 int sz, int sy, int sx) {
  const int v = blockIdx.x * blockDim.x + threadIdx.x;
  const int nv = Do * Ho * Wo;
  if (v >= nv) return;
  const int xo = v % Wo; int t = v / Wo; const int yo = t % Ho, zo = t / Ho;
  float r = 0.f;
  for (int kd = 0; kd < 3; ++kd)
    for (int kh = 0; kh < 3; ++kh)
      for (int kw = 0; kw < 3; ++kw) {
        const int zi = zo * sz - 1 + kd, yi = yo * sy - 1 + kh, xi = xo * sx - 1 + kw;
        if (zi >= 0 && zi < Di && yi >= 0 && yi < Hi && xi >= 0 && xi < Wi)
          r += mi[((size_t)zi * Hi + yi) * Wi + xi];
      }
  mo[v] = (r > 0.f) ? 1.f : 0.f;
}

__global__ void mul_kernel(const _Float16* __restrict__ a, const _Float16* __restrict__ b,
                           _Float16* __restrict__ o, size_t n) {
  const size_t i = (size_t)blockIdx.x * blockDim.x + threadIdx.x;
  if (i < n) o[i] = (_Float16)((float)a[i] * (float)b[i]);
}

// NDHWC(20) f16 -> NCDHW fp32 output transpose
__global__ void transpose_out_kernel(const _Float16* __restrict__ src,
                                     float* __restrict__ dst, int nvox, int C) {
  const size_t i = (size_t)blockIdx.x * blockDim.x + threadIdx.x;
  if (i >= (size_t)nvox * C) return;
  const int cls = (int)(i / nvox);
  const int vox = (int)(i - (size_t)cls * nvox);
  dst[i] = (float)src[(size_t)vox * C + cls];
}

// ---------------------------------------------------------------------------
// Host launch helpers
// ---------------------------------------------------------------------------
template <int KD, int KH, int KW, int CI, int CSPLIT = CI>
static inline void conv_t(hipStream_t st, const _Float16* in, const _Float16* in2,
                          int Di, int Hi, int Wi, const _Float16* wt,
                          _Float16* out, int Do, int Ho, int Wo, int Co,
                          int sz, int sy, int sx, int pz, int py, int px,
                          int dz, int dy, int dx,
                          const float* mask, const _Float16* add, const float* bias,
                          int act, float scale) {
  const int CoP = (Co + 31) & ~31;
  dim3 grid((unsigned)((Do * Ho * Wo + 15) / 16), (unsigned)(CoP / 32), 1);
  conv_wmma_t<KD, KH, KW, CI, CSPLIT><<<grid, dim3(32, 1, 1), 0, st>>>(
      in, in2, Di, Hi, Wi, wt, CoP, out, Do, Ho, Wo, Co,
      sz, sy, sx, pz, py, px, dz, dy, dx, mask, add, bias, act, scale);
}

// stride-1 SAME submanifold conv with fused mask*BN + activation (+ optional add)
template <int KD, int KH, int KW, int CI>
static inline void subm_t(hipStream_t st, const _Float16* in, int Di, int Hi, int Wi,
                          const _Float16* wt, int Co, _Float16* out,
                          const float* mask, const _Float16* add, int act) {
  conv_t<KD, KH, KW, CI>(st, in, in, Di, Hi, Wi, wt, out, Di, Hi, Wi, Co,
                         1, 1, 1, (KD - 1) / 2, (KH - 1) / 2, (KW - 1) / 2,
                         1, 1, 1, mask, add, nullptr, act, BN_SCALE);
}

// ---------------------------------------------------------------------------
extern "C" void kernel_launch(void* const* d_in, const int* in_sizes, int n_in,
                              void* d_out, int out_size, void* d_ws, size_t ws_size,
                              hipStream_t stream) {
  (void)n_in; (void)ws_size; (void)out_size;
  const float* feats = (const float*)d_in[0];
  const int*   coors = (const int*)d_in[1];
  // d_in[2] = batch_size (always 1)

  // params flattened in jax pytree order (sorted dict keys, DFS):
  const float* ctx_c1a = (const float*)d_in[3];   // (1,3,3,16,16)
  const float* ctx_c1b = (const float*)d_in[4];   // (3,1,3,16,16)
  const float* ctx_c2a = (const float*)d_in[5];   // (3,1,3,16,16)
  const float* ctx_c2b = (const float*)d_in[6];   // (1,3,3,16,16)
  const float* logits_b = (const float*)d_in[7];  // (20,)
  const float* logits_w = (const float*)d_in[8];  // (3,3,3,64,20)
  const float* rec_r1 = (const float*)d_in[9];    // (3,1,1,32,32)
  const float* rec_r2 = (const float*)d_in[10];   // (1,3,1,32,32)
  const float* rec_r3 = (const float*)d_in[11];   // (1,1,3,32,32)
  const float* r2_c1a = (const float*)d_in[12]; const float* r2_c1b = (const float*)d_in[13];
  const float* r2_c2a = (const float*)d_in[14]; const float* r2_c2b = (const float*)d_in[15];
  const float* r2_pool = (const float*)d_in[16];
  const float* r3_c1a = (const float*)d_in[17]; const float* r3_c1b = (const float*)d_in[18];
  const float* r3_c2a = (const float*)d_in[19]; const float* r3_c2b = (const float*)d_in[20];
  const float* r3_pool = (const float*)d_in[21];
  const float* r4_c1a = (const float*)d_in[22]; const float* r4_c1b = (const float*)d_in[23];
  const float* r4_c2a = (const float*)d_in[24]; const float* r4_c2b = (const float*)d_in[25];
  const float* r4_pool = (const float*)d_in[26];
  const float* r5_c1a = (const float*)d_in[27]; const float* r5_c1b = (const float*)d_in[28];
  const float* r5_c2a = (const float*)d_in[29]; const float* r5_c2b = (const float*)d_in[30];
  const float* r5_pool = (const float*)d_in[31];
  const float* u0_c1 = (const float*)d_in[32]; const float* u0_c2 = (const float*)d_in[33];
  const float* u0_c3 = (const float*)d_in[34]; const float* u0_tr = (const float*)d_in[35];
  const float* u0_up = (const float*)d_in[36];
  const float* u1_c1 = (const float*)d_in[37]; const float* u1_c2 = (const float*)d_in[38];
  const float* u1_c3 = (const float*)d_in[39]; const float* u1_tr = (const float*)d_in[40];
  const float* u1_up = (const float*)d_in[41];
  const float* u2_c1 = (const float*)d_in[42]; const float* u2_c2 = (const float*)d_in[43];
  const float* u2_c3 = (const float*)d_in[44]; const float* u2_tr = (const float*)d_in[45];
  const float* u2_up = (const float*)d_in[46];
  const float* u3_c1 = (const float*)d_in[47]; const float* u3_c2 = (const float*)d_in[48];
  const float* u3_c3 = (const float*)d_in[49]; const float* u3_tr = (const float*)d_in[50];
  const float* u3_up = (const float*)d_in[51];

  const int NV = in_sizes[0] / 16;
  const int D0 = 128, H0 = 96, W0 = 16; const int n0 = D0 * H0 * W0;
  const int D1 = 64,  H1 = 48, W1 = 8;  const int n1 = D1 * H1 * W1;
  const int D2 = 32,  H2 = 24, W2 = 4;  const int n2 = D2 * H2 * W2;
  const int D3 = 16,  H3 = 12, W3 = 4;  const int n3 = D3 * H3 * W3;
  const int D4 = 8,   H4 = 6,  W4 = 4;  const int n4 = D4 * H4 * W4;

  // ---- bump-arena over d_ws (f16 activations: whole net ~105MB, L2-resident)
  size_t off = 0;
  char* abase = (char*)d_ws;
  auto AL = [&](size_t nf) -> float* {
    float* p = (float*)(abase + off);
    off = (off + nf * sizeof(float) + 255) & ~(size_t)255;
    return p;
  };
  auto ALH = [&](size_t nh) -> _Float16* {
    _Float16* p = (_Float16*)(abase + off);
    off = (off + nh * sizeof(_Float16) + 255) & ~(size_t)255;
    return p;
  };
  float* m0 = AL(n0); float* m1 = AL(n1); float* m2 = AL(n2);
  float* m3 = AL(n3); float* m4 = AL(n4);
  _Float16* X   = ALH((size_t)n0 * 16);
  _Float16* T16 = ALH((size_t)n0 * 16);
  _Float16* S16 = ALH((size_t)n0 * 16);
  _Float16* T32 = ALH((size_t)n0 * 32);   // later reused as recon gate R
  _Float16* S32 = ALH((size_t)n0 * 32);   // later reused as 20-ch logits scratch
  _Float16* D1B = ALH((size_t)n0 * 32);
  _Float16* UA0 = ALH((size_t)n0 * 32);
  _Float16* UB0 = ALH((size_t)n0 * 32);
  _Float16* A1  = ALH((size_t)n1 * 32);   // d1c; later reused as up3 trans output
  _Float16* T64 = ALH((size_t)n1 * 64);
  _Float16* S64 = ALH((size_t)n1 * 64);
  _Float16* D2B = ALH((size_t)n1 * 64);
  _Float16* UA1 = ALH((size_t)n1 * 64);
  _Float16* UB1 = ALH((size_t)n1 * 64);
  _Float16* B2  = ALH((size_t)n2 * 64);   // d2c; later reused as up2 trans output
  _Float16* T128 = ALH((size_t)n2 * 128);
  _Float16* S128 = ALH((size_t)n2 * 128);
  _Float16* D3B  = ALH((size_t)n2 * 128);
  _Float16* UA2  = ALH((size_t)n2 * 128);
  _Float16* UB2  = ALH((size_t)n2 * 128);
  _Float16* B3   = ALH((size_t)n3 * 128); // d3c; later reused as up1 trans output
  _Float16* T256 = ALH((size_t)n3 * 256);
  _Float16* S256 = ALH((size_t)n3 * 256);
  _Float16* D4B  = ALH((size_t)n3 * 256);
  _Float16* UA3  = ALH((size_t)n3 * 256);
  _Float16* UB3  = ALH((size_t)n3 * 256);
  _Float16* B4   = ALH((size_t)n4 * 256); // d4c
  _Float16* C4   = ALH((size_t)n4 * 256);

  // ---- weight prep: f16 [tap][co_pad32][ci] (+flip for transposed convs)
  auto prep = [&](const float* w, int taps, int Ci, int Co, int flip) -> const _Float16* {
    const int tapsPad = (Ci == 16) ? ((taps + 1) & ~1) : taps;
    const int CoP = (Co + 31) & ~31;
    const size_t total = (size_t)tapsPad * CoP * Ci;
    _Float16* wt = ALH(total);
    wprep_kernel<<<(int)((total + 255) / 256), 256, 0, stream>>>(
        w, wt, taps, tapsPad, Ci, Co, CoP, flip);
    return wt;
  };
  const _Float16* W_ctx_c1a = prep(ctx_c1a, 9, 16, 16, 0);
  const _Float16* W_ctx_c1b = prep(ctx_c1b, 9, 16, 16, 0);
  const _Float16* W_ctx_c2a = prep(ctx_c2a, 9, 16, 16, 0);
  const _Float16* W_ctx_c2b = prep(ctx_c2b, 9, 16, 16, 0);
  const _Float16* W_r2_c1a = prep(r2_c1a, 9, 16, 32, 0);
  const _Float16* W_r2_c1b = prep(r2_c1b, 9, 32, 32, 0);
  const _Float16* W_r2_c2a = prep(r2_c2a, 9, 16, 32, 0);
  const _Float16* W_r2_c2b = prep(r2_c2b, 9, 32, 32, 0);
  const _Float16* W_r2_pool = prep(r2_pool, 27, 32, 32, 0);
  const _Float16* W_r3_c1a = prep(r3_c1a, 9, 32, 64, 0);
  const _Float16* W_r3_c1b = prep(r3_c1b, 9, 64, 64, 0);
  const _Float16* W_r3_c2a = prep(r3_c2a, 9, 32, 64, 0);
  const _Float16* W_r3_c2b = prep(r3_c2b, 9, 64, 64, 0);
  const _Float16* W_r3_pool = prep(r3_pool, 27, 64, 64, 0);
  const _Float16* W_r4_c1a = prep(r4_c1a, 9, 64, 128, 0);
  const _Float16* W_r4_c1b = prep(r4_c1b, 9, 128, 128, 0);
  const _Float16* W_r4_c2a = prep(r4_c2a, 9, 64, 128, 0);
  const _Float16* W_r4_c2b = prep(r4_c2b, 9, 128, 128, 0);
  const _Float16* W_r4_pool = prep(r4_pool, 27, 128, 128, 0);
  const _Float16* W_r5_c1a = prep(r5_c1a, 9, 128, 256, 0);
  const _Float16* W_r5_c1b = prep(r5_c1b, 9, 256, 256, 0);
  const _Float16* W_r5_c2a = prep(r5_c2a, 9, 128, 256, 0);
  const _Float16* W_r5_c2b = prep(r5_c2b, 9, 256, 256, 0);
  const _Float16* W_r5_pool = prep(r5_pool, 27, 256, 256, 0);
  const _Float16* W_u0_c1 = prep(u0_c1, 9, 256, 256, 0);
  const _Float16* W_u0_c2 = prep(u0_c2, 9, 256, 256, 0);
  const _Float16* W_u0_c3 = prep(u0_c3, 27, 256, 256, 0);
  const _Float16* W_u0_tr = prep(u0_tr, 27, 256, 256, 0);
  const _Float16* W_u0_up = prep(u0_up, 27, 256, 256, 1); // flipped (transposed conv)
  const _Float16* W_u1_c1 = prep(u1_c1, 9, 128, 128, 0);
  const _Float16* W_u1_c2 = prep(u1_c2, 9, 128, 128, 0);
  const _Float16* W_u1_c3 = prep(u1_c3, 27, 128, 128, 0);
  const _Float16* W_u1_tr = prep(u1_tr, 27, 256, 128, 0);
  const _Float16* W_u1_up = prep(u1_up, 27, 128, 128, 1);
  const _Float16* W_u2_c1 = prep(u2_c1, 9, 64, 64, 0);
  const _Float16* W_u2_c2 = prep(u2_c2, 9, 64, 64, 0);
  const _Float16* W_u2_c3 = prep(u2_c3, 27, 64, 64, 0);
  const _Float16* W_u2_tr = prep(u2_tr, 27, 128, 64, 0);
  const _Float16* W_u2_up = prep(u2_up, 27, 64, 64, 1);
  const _Float16* W_u3_c1 = prep(u3_c1, 9, 32, 32, 0);
  const _Float16* W_u3_c2 = prep(u3_c2, 9, 32, 32, 0);
  const _Float16* W_u3_c3 = prep(u3_c3, 27, 32, 32, 0);
  const _Float16* W_u3_tr = prep(u3_tr, 27, 64, 32, 0);
  const _Float16* W_u3_up = prep(u3_up, 27, 32, 32, 1);
  const _Float16* W_r1 = prep(rec_r1, 3, 32, 32, 0);
  const _Float16* W_r2 = prep(rec_r2, 3, 32, 32, 0);
  const _Float16* W_r3w = prep(rec_r3, 3, 32, 32, 0);
  const _Float16* W_log = prep(logits_w, 27, 64, 20, 0);

  // ---- densify: zero + scatter (X is f16: n0*16 halves == n0*8 floats)
  zero_kernel<<<1024, 256, 0, stream>>>((float*)X, (size_t)n0 * 8);
  zero_kernel<<<512, 256, 0, stream>>>(m0, (size_t)n0);
  scatter_kernel<<<(NV * 16 + 255) / 256, 256, 0, stream>>>(feats, coors, NV, 16, X, m0, H0, W0);

  // ---- mask pyramid
  mask_down_kernel<<<(n1 + 255) / 256, 256, 0, stream>>>(m0, D0, H0, W0, m1, D1, H1, W1, 2, 2, 2);
  mask_down_kernel<<<(n2 + 255) / 256, 256, 0, stream>>>(m1, D1, H1, W1, m2, D2, H2, W2, 2, 2, 2);
  mask_down_kernel<<<(n3 + 255) / 256, 256, 0, stream>>>(m2, D2, H2, W2, m3, D3, H3, W3, 2, 2, 1);
  mask_down_kernel<<<(n4 + 255) / 256, 256, 0, stream>>>(m3, D3, H3, W3, m4, D4, H4, W4, 2, 2, 1);

  // ---- context block (16 -> 16), x1 written back into X
  subm_t<1, 3, 3, 16>(stream, X,   D0, H0, W0, W_ctx_c1a, 16, T16, m0, nullptr, 1);
  subm_t<3, 1, 3, 16>(stream, T16, D0, H0, W0, W_ctx_c1b, 16, S16, m0, nullptr, 1);
  subm_t<3, 1, 3, 16>(stream, X,   D0, H0, W0, W_ctx_c2a, 16, T16, m0, nullptr, 1);
  subm_t<1, 3, 3, 16>(stream, T16, D0, H0, W0, W_ctx_c2b, 16, X,   m0, S16,     1);

  // ---- res2 (16 -> 32) @L0, pool stride (2,2,2) -> A1
  subm_t<3, 1, 3, 16>(stream, X,   D0, H0, W0, W_r2_c1a, 32, T32, m0, nullptr, 1);
  subm_t<1, 3, 3, 32>(stream, T32, D0, H0, W0, W_r2_c1b, 32, S32, m0, nullptr, 1);
  subm_t<1, 3, 3, 16>(stream, X,   D0, H0, W0, W_r2_c2a, 32, T32, m0, nullptr, 1);
  subm_t<3, 1, 3, 32>(stream, T32, D0, H0, W0, W_r2_c2b, 32, D1B, m0, S32,     1);
  conv_t<3, 3, 3, 32>(stream, D1B, D1B, D0, H0, W0, W_r2_pool, A1, D1, H1, W1, 32,
                      2, 2, 2, 1, 1, 1, 1, 1, 1, nullptr, nullptr, nullptr, 0, 1.f);

  // ---- res3 (32 -> 64) @L1, pool (2,2,2) -> B2
  subm_t<3, 1, 3, 32>(stream, A1,  D1, H1, W1, W_r3_c1a, 64, T64, m1, nullptr, 1);
  subm_t<1, 3, 3, 64>(stream, T64, D1, H1, W1, W_r3_c1b, 64, S64, m1, nullptr, 1);
  subm_t<1, 3, 3, 32>(stream, A1,  D1, H1, W1, W_r3_c2a, 64, T64, m1, nullptr, 1);
  subm_t<3, 1, 3, 64>(stream, T64, D1, H1, W1, W_r3_c2b, 64, D2B, m1, S64,     1);
  conv_t<3, 3, 3, 64>(stream, D2B, D2B, D1, H1, W1, W_r3_pool, B2, D2, H2, W2, 64,
                      2, 2, 2, 1, 1, 1, 1, 1, 1, nullptr, nullptr, nullptr, 0, 1.f);

  // ---- res4 (64 -> 128) @L2, pool (2,2,1) -> B3
  subm_t<3, 1, 3, 64>(stream, B2,    D2, H2, W2, W_r4_c1a, 128, T128, m2, nullptr, 1);
  subm_t<1, 3, 3, 128>(stream, T128, D2, H2, W2, W_r4_c1b, 128, S128, m2, nullptr, 1);
  subm_t<1, 3, 3, 64>(stream, B2,    D2, H2, W2, W_r4_c2a, 128, T128, m2, nullptr, 1);
  subm_t<3, 1, 3, 128>(stream, T128, D2, H2, W2, W_r4_c2b, 128, D3B,  m2, S128,   1);
  conv_t<3, 3, 3, 128>(stream, D3B, D3B, D2, H2, W2, W_r4_pool, B3, D3, H3, W3, 128,
                       2, 2, 1, 1, 1, 1, 1, 1, 1, nullptr, nullptr, nullptr, 0, 1.f);

  // ---- res5 (128 -> 256) @L3, pool (2,2,1) -> B4
  subm_t<3, 1, 3, 128>(stream, B3,   D3, H3, W3, W_r5_c1a, 256, T256, m3, nullptr, 1);
  subm_t<1, 3, 3, 256>(stream, T256, D3, H3, W3, W_r5_c1b, 256, S256, m3, nullptr, 1);
  subm_t<1, 3, 3, 128>(stream, B3,   D3, H3, W3, W_r5_c2a, 256, T256, m3, nullptr, 1);
  subm_t<3, 1, 3, 256>(stream, T256, D3, H3, W3, W_r5_c2b, 256, D4B,  m3, S256,   1);
  conv_t<3, 3, 3, 256>(stream, D4B, D4B, D3, H3, W3, W_r5_pool, B4, D4, H4, W4, 256,
                       2, 2, 1, 1, 1, 1, 1, 1, 1, nullptr, nullptr, nullptr, 0, 1.f);

  // ---- up0: L4(256) -> L3(256); transposed conv = flipped weights + lhs-dilation
  subm_t<3, 3, 3, 256>(stream, B4, D4, H4, W4, W_u0_tr, 256, C4, m4, nullptr, 1);
  conv_t<3, 3, 3, 256>(stream, C4, C4, D4, H4, W4, W_u0_up, UA3, D3, H3, W3, 256,
                       1, 1, 1, 1, 1, 1, 2, 2, 1, m3, D4B, nullptr, 0, 1.f);
  subm_t<1, 3, 3, 256>(stream, UA3, D3, H3, W3, W_u0_c1, 256, UB3, m3, nullptr, 1);
  subm_t<3, 1, 3, 256>(stream, UB3, D3, H3, W3, W_u0_c2, 256, UA3, m3, nullptr, 1);
  subm_t<3, 3, 3, 256>(stream, UA3, D3, H3, W3, W_u0_c3, 256, UB3, m3, nullptr, 1); // u4

  // ---- up1: L3(256) -> L2(128)   (B3 reused for trans output)
  subm_t<3, 3, 3, 256>(stream, UB3, D3, H3, W3, W_u1_tr, 128, B3, m3, nullptr, 1);
  conv_t<3, 3, 3, 128>(stream, B3, B3, D3, H3, W3, W_u1_up, UA2, D2, H2, W2, 128,
                       1, 1, 1, 1, 1, 1, 2, 2, 1, m2, D3B, nullptr, 0, 1.f);
  subm_t<1, 3, 3, 128>(stream, UA2, D2, H2, W2, W_u1_c1, 128, UB2, m2, nullptr, 1);
  subm_t<3, 1, 3, 128>(stream, UB2, D2, H2, W2, W_u1_c2, 128, UA2, m2, nullptr, 1);
  subm_t<3, 3, 3, 128>(stream, UA2, D2, H2, W2, W_u1_c3, 128, UB2, m2, nullptr, 1); // u3

  // ---- up2: L2(128) -> L1(64)   (B2 reused for trans output)
  subm_t<3, 3, 3, 128>(stream, UB2, D2, H2, W2, W_u2_tr, 64, B2, m2, nullptr, 1);
  conv_t<3, 3, 3, 64>(stream, B2, B2, D2, H2, W2, W_u2_up, UA1, D1, H1, W1, 64,
                      1, 1, 1, 1, 1, 1, 2, 2, 2, m1, D2B, nullptr, 0, 1.f);
  subm_t<1, 3, 3, 64>(stream, UA1, D1, H1, W1, W_u2_c1, 64, UB1, m1, nullptr, 1);
  subm_t<3, 1, 3, 64>(stream, UB1, D1, H1, W1, W_u2_c2, 64, UA1, m1, nullptr, 1);
  subm_t<3, 3, 3, 64>(stream, UA1, D1, H1, W1, W_u2_c3, 64, UB1, m1, nullptr, 1); // u2

  // ---- up3: L1(64) -> L0(32)   (A1 reused for trans output)
  subm_t<3, 3, 3, 64>(stream, UB1, D1, H1, W1, W_u3_tr, 32, A1, m1, nullptr, 1);
  conv_t<3, 3, 3, 32>(stream, A1, A1, D1, H1, W1, W_u3_up, UA0, D0, H0, W0, 32,
                      1, 1, 1, 1, 1, 1, 2, 2, 2, m0, D1B, nullptr, 0, 1.f);
  subm_t<1, 3, 3, 32>(stream, UA0, D0, H0, W0, W_u3_c1, 32, UB0, m0, nullptr, 1);
  subm_t<3, 1, 3, 32>(stream, UB0, D0, H0, W0, W_u3_c2, 32, UA0, m0, nullptr, 1);
  subm_t<3, 3, 3, 32>(stream, UA0, D0, H0, W0, W_u3_c3, 32, UB0, m0, nullptr, 1); // u1

  // ---- recon gate: R = sig(r1) + sig(r2) + sig(r3)  (R = T32, in-place adds)
  subm_t<3, 1, 1, 32>(stream, UB0, D0, H0, W0, W_r1,  32, T32, m0, nullptr, 2);
  subm_t<1, 3, 1, 32>(stream, UB0, D0, H0, W0, W_r2,  32, T32, m0, T32,     2);
  subm_t<1, 1, 3, 32>(stream, UB0, D0, H0, W0, W_r3w, 32, T32, m0, T32,     2);
  // u0 = R * u1 -> UA0
  mul_kernel<<<(int)(((size_t)n0 * 32 + 255) / 256), 256, 0, stream>>>(
      T32, UB0, UA0, (size_t)n0 * 32);

  // ---- logits over virtual concat [u0 (UA0) | u1 (UB0)]: CI=64, CSPLIT=32
  conv_t<3, 3, 3, 64, 32>(stream, UA0, UB0, D0, H0, W0, W_log, S32, D0, H0, W0, 20,
                          1, 1, 1, 1, 1, 1, 1, 1, 1, m0, nullptr, logits_b, 0, 1.f);

  // ---- NDHWC f16 -> NCDHW fp32
  transpose_out_kernel<<<(int)(((size_t)n0 * 20 + 255) / 256), 256, 0, stream>>>(
      S32, (float*)d_out, n0, 20);
}
